// MPS_pytorch_sharing_output_83880711290967
// MI455X (gfx1250) — matrix operand order repaired and checked by reading, not compiled, not run on hardware
//
#include <hip/hip_runtime.h>

typedef __attribute__((ext_vector_type(16))) __bf16   v16bf;
typedef __attribute__((ext_vector_type(8)))  float    v8f;
typedef __attribute__((ext_vector_type(8)))  unsigned v8u;
typedef __attribute__((ext_vector_type(4)))  unsigned u32x4;

#define NROWS_TOTAL 3136          // B*P = 4*784
#define DD 64                     // bond/feature dim
#define L_SITES 421               // contracted MPS sites
#define SITE_STRIDE 27840         // H*W*D = 15*29*64 floats per sample row
#define T_STRIDE (64*64*64)       // D^3 floats per site tensor
#define WG_ROWS 64                // rows of N per workgroup
#define NBLOCKS (NROWS_TOTAL / WG_ROWS)   // 49

__device__ __forceinline__ float4 ldg4(const float* p) {
  return *reinterpret_cast<const float4*>(p);
}

// pack two floats into one VGPR as bf16 pair (lo = first, hi = second)
__device__ __forceinline__ unsigned bfpair(float lo, float hi) {
  unsigned short l = __builtin_bit_cast(unsigned short, (__bf16)lo);
  unsigned short h = __builtin_bit_cast(unsigned short, (__bf16)hi);
  return (unsigned)l | ((unsigned)h << 16);
}

// CDNA5 packed bf16 multiply (VOP3P V_PK_MUL_BF16)
__device__ __forceinline__ unsigned pkmul_bf16(unsigned a, unsigned b) {
  unsigned d;
  asm("v_pk_mul_bf16 %0, %1, %2" : "=v"(d) : "v"(a), "v"(b));
  return d;
}

// async copy 16 bytes global -> LDS (ASYNCcnt-tracked, CDNA5)
__device__ __forceinline__ void async_ld16(unsigned lds_off, const float* gptr) {
  asm volatile("global_load_async_to_lds_b128 %0, %1, off"
               :: "v"(lds_off), "v"((unsigned long long)(uintptr_t)gptr)
               : "memory");
}

// load 16 consecutive floats from global, convert to bf16 fragment (B operand)
__device__ __forceinline__ v16bf cvt16(const float* p) {
  float4 b0 = ldg4(p), b1 = ldg4(p + 4), b2 = ldg4(p + 8), b3 = ldg4(p + 12);
  v16bf r;
  r[ 0]=(__bf16)b0.x; r[ 1]=(__bf16)b0.y; r[ 2]=(__bf16)b0.z; r[ 3]=(__bf16)b0.w;
  r[ 4]=(__bf16)b1.x; r[ 5]=(__bf16)b1.y; r[ 6]=(__bf16)b1.z; r[ 7]=(__bf16)b1.w;
  r[ 8]=(__bf16)b2.x; r[ 9]=(__bf16)b2.y; r[10]=(__bf16)b2.z; r[11]=(__bf16)b2.w;
  r[12]=(__bf16)b3.x; r[13]=(__bf16)b3.y; r[14]=(__bf16)b3.z; r[15]=(__bf16)b3.w;
  return r;
}

#define WMMA_BF16(A, B, C) \
  __builtin_amdgcn_wmma_f32_16x16x32_bf16(false, (A), false, (B), (short)0, (C), false, false)

__global__ __launch_bounds__(256)
void mps_wmma_kernel(const float* __restrict__ samples,
                     const float* __restrict__ tensors,
                     const float* __restrict__ tensors2,
                     float* __restrict__ out)
{
  __shared__ float    lds_left[WG_ROWS][65];     // f32 state, padded
  __shared__ unsigned lds_sbf[WG_ROWS][36];      // s_i as packed bf16 pairs
  __shared__ float    lds_sraw[2][WG_ROWS][68];  // async landing buffers (f32)

  const int tid  = threadIdx.x;
  const int wid  = tid >> 5;        // 8 waves (wave32)
  const int lane = tid & 31;
  const int n0   = blockIdx.x * WG_ROWS;

  const int kg  = wid >> 2;         // K split: l in [kg*32, kg*32+32)
  const int cp  = (wid >> 1) & 1;   // col pair: cols [cp*32, cp*32+32)
  const int rgp = wid & 1;          // row group: rows [rgp*32, rgp*32+32)

  const int m   = lane & 15;        // A M-row within tile / B col within tile
  const int hi  = lane >> 4;        // lane half (K split: half hi holds K 16*hi..16*hi+15)
  const int col = lane & 15;        // C/D column within tile

  // init left = e0
  for (int k = tid; k < WG_ROWS * DD; k += 256)
    lds_left[k >> 6][k & 63] = ((k & 63) == 0) ? 1.0f : 0.0f;

  // s staging geometry: 4 threads per row, 16 floats each
  const int srow = tid >> 2;
  const int sq   = tid & 3;
  const float* srow_base = samples + (size_t)(n0 + srow) * SITE_STRIDE + sq * 16;

  // prime the pipeline: async-prefetch s_0 into buffer 0
  {
    const float* src = srow_base;   // i = 0
    unsigned dst = (unsigned)(uintptr_t)&lds_sraw[0][srow][sq * 16];
    async_ld16(dst +  0, src + 0);
    async_ld16(dst + 16, src + 4);
    async_ld16(dst + 32, src + 8);
    async_ld16(dst + 48, src + 12);
  }

  for (int i = 0; i < L_SITES; ++i) {
    const int buf = i & 1;

    asm volatile("s_wait_asynccnt 0" ::: "memory");  // own prefetch landed
    __syncthreads();  // (a) all prefetch writes visible; previous step retired

    { // pack raw f32 s-tile -> packed bf16
      const float4* rp =
          reinterpret_cast<const float4*>(&lds_sraw[buf][srow][sq * 16]);
      float4 a0 = rp[0], a1 = rp[1], a2 = rp[2], a3 = rp[3];
      u32x4 p0, p1;
      p0.x = bfpair(a0.x, a0.y); p0.y = bfpair(a0.z, a0.w);
      p0.z = bfpair(a1.x, a1.y); p0.w = bfpair(a1.z, a1.w);
      p1.x = bfpair(a2.x, a2.y); p1.y = bfpair(a2.z, a2.w);
      p1.z = bfpair(a3.x, a3.y); p1.w = bfpair(a3.z, a3.w);
      u32x4* dst = reinterpret_cast<u32x4*>(&lds_sbf[srow][sq * 8]);
      dst[0] = p0; dst[1] = p1;
    }

    { // async-prefetch s_{i+1} into the other buffer (stays in-bounds: 421 < 435)
      const float* src = srow_base + (size_t)(i + 1) * DD;
      unsigned dst = (unsigned)(uintptr_t)&lds_sraw[buf ^ 1][srow][sq * 16];
      async_ld16(dst +  0, src + 0);
      async_ld16(dst + 16, src + 4);
      async_ld16(dst + 32, src + 8);
      async_ld16(dst + 48, src + 12);
    }
    __syncthreads();  // (b) packed s ready, left stable

    // accumulators: [rt][ct] over rows rgp*32 + rt*16, cols cp*32 + ct*16
    v8f acc00 = {}, acc01 = {}, acc10 = {}, acc11 = {};
    if (kg == 0) {  // identity (+left) term carried by the kg=0 waves
#pragma unroll
      for (int v = 0; v < 8; ++v) {
        acc00[v] = lds_left[rgp*32 +  0 + hi*8 + v][cp*32 +  0 + col];
        acc01[v] = lds_left[rgp*32 +  0 + hi*8 + v][cp*32 + 16 + col];
        acc10[v] = lds_left[rgp*32 + 16 + hi*8 + v][cp*32 +  0 + col];
        acc11[v] = lds_left[rgp*32 + 16 + hi*8 + v][cp*32 + 16 + col];
      }
    }

    const float* Ti  = tensors + (size_t)i * T_STRIDE;
    const float* tb0 = Ti + (size_t)(cp*32 +  0 + col) * DD + hi * 16;
    const float* tb1 = Ti + (size_t)(cp*32 + 16 + col) * DD + hi * 16;

#pragma unroll 2
    for (int l = kg * 32; l < kg * 32 + 32; ++l) {
      const float lf0 = lds_left[rgp*32 +  0 + m][l];
      const float lf1 = lds_left[rgp*32 + 16 + m][l];
      const unsigned lml0 = bfpair(lf0, lf0);   // broadcast bf16 pair
      const unsigned lml1 = bfpair(lf1, lf1);
      const float* t0 = tb0 + l * (DD * DD);
      const float* t1 = tb1 + l * (DD * DD);

#pragma unroll
      for (int half = 0; half < 2; ++half) {
        const int d0 = half << 5;   // K chunk covers d in [d0, d0+32)

        // ---- B fragments (two col tiles), f32 global -> bf16
        v16bf B0 = cvt16(t0 + d0);
        v16bf B1 = cvt16(t1 + d0);

        // ---- packed-bf16 s slices (shared by both row tiles)
        // A layout: lanes 0-15 hold K {0..15}; lanes 16-31 {16..31}
        const u32x4* sp = reinterpret_cast<const u32x4*>(&lds_sbf[m][0]);
        const u32x4 sA = sp[(d0 >> 3) + 2 * hi];        // d pairs b1..b1+7
        const u32x4 sB = sp[(d0 >> 3) + 2 * hi + 1];    // d pairs b1+8..b1+15

        // ---- A fragments via v_pk_mul_bf16
        v8u a0u, a1u;
        a0u[0]=pkmul_bf16(lml0,sA.x); a0u[1]=pkmul_bf16(lml0,sA.y);
        a0u[2]=pkmul_bf16(lml0,sA.z); a0u[3]=pkmul_bf16(lml0,sA.w);
        a0u[4]=pkmul_bf16(lml0,sB.x); a0u[5]=pkmul_bf16(lml0,sB.y);
        a0u[6]=pkmul_bf16(lml0,sB.z); a0u[7]=pkmul_bf16(lml0,sB.w);
        a1u[0]=pkmul_bf16(lml1,sA.x); a1u[1]=pkmul_bf16(lml1,sA.y);
        a1u[2]=pkmul_bf16(lml1,sA.z); a1u[3]=pkmul_bf16(lml1,sA.w);
        a1u[4]=pkmul_bf16(lml1,sB.x); a1u[5]=pkmul_bf16(lml1,sB.y);
        a1u[6]=pkmul_bf16(lml1,sB.z); a1u[7]=pkmul_bf16(lml1,sB.w);
        const v16bf A0 = __builtin_bit_cast(v16bf, a0u);
        const v16bf A1 = __builtin_bit_cast(v16bf, a1u);

        // ---- 4 independent WMMAs
        acc00 = WMMA_BF16(A0, B0, acc00);
        acc01 = WMMA_BF16(A0, B1, acc01);
        acc10 = WMMA_BF16(A1, B0, acc10);
        acc11 = WMMA_BF16(A1, B1, acc11);
      }
    }

    __syncthreads();  // (c) all reads of lds_left done

    if (kg == 0) {    // write partial (identity + l<32) state
#pragma unroll
      for (int v = 0; v < 8; ++v) {
        lds_left[rgp*32 +  0 + hi*8 + v][cp*32 +  0 + col] = acc00[v];
        lds_left[rgp*32 +  0 + hi*8 + v][cp*32 + 16 + col] = acc01[v];
        lds_left[rgp*32 + 16 + hi*8 + v][cp*32 +  0 + col] = acc10[v];
        lds_left[rgp*32 + 16 + hi*8 + v][cp*32 + 16 + col] = acc11[v];
      }
    }
    __syncthreads();  // (d) kg=0 write-back visible

    if (kg == 1) {    // merge l>=32 partial sums
#pragma unroll
      for (int v = 0; v < 8; ++v) {
        lds_left[rgp*32 +  0 + hi*8 + v][cp*32 +  0 + col] += acc00[v];
        lds_left[rgp*32 +  0 + hi*8 + v][cp*32 + 16 + col] += acc01[v];
        lds_left[rgp*32 + 16 + hi*8 + v][cp*32 +  0 + col] += acc10[v];
        lds_left[rgp*32 + 16 + hi*8 + v][cp*32 + 16 + col] += acc11[v];
      }
    }
  }

  __syncthreads();

  // Epilogue: logits = left @ (tensors2[:,0,:] + e0 bias), then log_softmax
  if (tid < WG_ROWS) {
    const int row = tid;
    float a0 = 0.0f, a1 = 0.0f;
#pragma unroll 4
    for (int d = 0; d < DD; ++d) {
      const float lv   = lds_left[row][d];
      const float bias = (d == 0) ? 1.0f : 0.0f;
      a0 += lv * (tensors2[d * 128 + 0] + bias);
      a1 += lv * (tensors2[d * 128 + 1] + bias);
    }
    const float mx  = fmaxf(a0, a1);
    const float lse = mx + logf(expf(a0 - mx) + expf(a1 - mx));
    const int n = n0 + row;
    out[n * 2 + 0] = a0 - lse;
    out[n * 2 + 1] = a1 - lse;
  }
}

extern "C" void kernel_launch(void* const* d_in, const int* in_sizes, int n_in,
                              void* d_out, int out_size, void* d_ws, size_t ws_size,
                              hipStream_t stream) {
  (void)in_sizes; (void)n_in; (void)out_size; (void)d_ws; (void)ws_size;
  const float* samples  = (const float*)d_in[0];
  const float* tensors  = (const float*)d_in[1];
  const float* tensors2 = (const float*)d_in[2];
  float* out = (float*)d_out;

  mps_wmma_kernel<<<NBLOCKS, 256, 0, stream>>>(samples, tensors, tensors2, out);
}
